// SAModule_9019431321932
// MI455X (gfx1250) — matrix operand order repaired and checked by reading, not compiled, not run on hardware
//
#include <hip/hip_runtime.h>

// ---------- types ----------
typedef __attribute__((ext_vector_type(16))) __bf16 v16bf;
typedef __attribute__((ext_vector_type(8)))  __bf16 v8bf;
typedef __attribute__((ext_vector_type(8)))  float  v8f;

union V16U { v16bf v; v8bf h[2]; };

static __device__ __forceinline__ v16bf pack16(v8bf lo, v8bf hi) {
    V16U u; u.h[0] = lo; u.h[1] = hi; return u.v;
}

static __device__ __forceinline__ v8f wmma_bf16(v16bf a, v16bf b, v8f c) {
    // D = A(16x32 bf16) * B(32x16 bf16) + C(16x16 f32)
    return __builtin_amdgcn_wmma_f32_16x16x32_bf16(
        false, a, false, b, (short)0, c, false, false);
}

// A-fragment (16x32 bf16): lane L holds row (L&15); K chunks at
// (L>>4)*8 and 16+(L>>4)*8 (per CDNA5 16-bit A layout).
static __device__ __forceinline__ v16bf load_afrag(const __bf16* base, int stride, int lane) {
    const __bf16* p = base + (size_t)(lane & 15) * stride;
    int hi8 = (lane >> 4) * 8;
    v8bf lo = *(const v8bf*)(p + hi8);
    v8bf hi = *(const v8bf*)(p + 16 + hi8);
    return pack16(lo, hi);
}

// B-fragment (32x16 bf16): lane L holds column (L&15); K chunks at
// (L>>4)*8 and 16+(L>>4)*8 (same K order as the A-fragment).
static __device__ __forceinline__ v16bf load_bfrag(const __bf16* base, int stride, int lane) {
    const __bf16* p = base + (size_t)(lane & 15) * stride + (lane >> 4) * 8;
    v8bf lo = *(const v8bf*)(p);
    v8bf hi = *(const v8bf*)(p + 16);
    return pack16(lo, hi);
}

// ---------- constants ----------
#define CIN   256
#define NPIX  4096          // 64*64
#define COUT  320           // 32 q + 32 k + 256 v
#define QOFF  0
#define KOFF  32
#define VOFF  64

// =====================================================================
// Kernel 1: pack weights/biases -> bf16 Wcat[320][256], f32 bcat[320]
// =====================================================================
__global__ void pack_kernel(const float* __restrict__ w1, const float* __restrict__ b1,
                            const float* __restrict__ w2, const float* __restrict__ b2,
                            const float* __restrict__ w3, const float* __restrict__ b3,
                            __bf16* __restrict__ Wc, float* __restrict__ bc) {
    int id = blockIdx.x * blockDim.x + threadIdx.x;   // 0..81919
    if (id < COUT * CIN) {
        int o = id >> 8, c = id & 255;
        float v = (o < 32) ? w1[o * CIN + c]
                : (o < 64) ? w2[(o - 32) * CIN + c]
                           : w3[(o - 64) * CIN + c];
        Wc[id] = (__bf16)v;
    }
    if (id < COUT) {
        bc[id] = (id < 32) ? b1[id] : (id < 64) ? b2[id - 32] : b3[id - 64];
    }
}

// =====================================================================
// Kernel 2: fused QKV projection.  y[b][pix][0..319] (bf16)
//   y = x^T * Wcat^T + bcat,  x = feat_map [B][256][4096]
// Block: 256 thr (8 waves). Each block: 64 pixels x 320 outputs.
// Waves: (row-tile 0..3) x (col-half 0..1), 10 col-tiles each.
// =====================================================================
#define XPAD 264            // 256 + 8 bf16 pad -> conflict-free A reads
__global__ void proj_kernel(const float* __restrict__ x,
                            const __bf16* __restrict__ Wc,
                            const float* __restrict__ bc,
                            __bf16* __restrict__ y) {
    extern __shared__ char smem[];
    __bf16* xs = (__bf16*)smem;                      // [64][XPAD]
    int blk = blockIdx.x;
    int b  = blk >> 6;                               // 4096/64 = 64 tiles/batch
    int n0 = (blk & 63) * 64;
    int tid = threadIdx.x;

    const float* xb = x + (size_t)b * CIN * NPIX + n0;
    // stage x^T tile (coalesced global reads, padded LDS rows)
#pragma unroll 4
    for (int i = 0; i < 64; ++i) {
        int idx = i * 256 + tid;                     // 0..16383
        int ch = idx >> 6, pix = idx & 63;
        xs[pix * XPAD + ch] = (__bf16)xb[(size_t)ch * NPIX + pix];
    }
    __syncthreads();

    int wave = tid >> 5, lane = tid & 31;
    int pbase = (wave & 3) * 16;
    int chalf = wave >> 2;

    // preload this wave's A-fragments (16 pixels x 256 ch = 8 k-steps)
    v16bf afr[8];
#pragma unroll
    for (int k = 0; k < 8; ++k)
        afr[k] = load_afrag(xs + pbase * XPAD + k * 32, XPAD, lane);

    __bf16* yb = y + (size_t)b * NPIX * COUT + (size_t)n0 * COUT;
    int col = lane & 15, rhi = (lane >> 4) * 8;

    for (int ct = 0; ct < 10; ++ct) {
        int ob = (chalf * 10 + ct) * 16;
        v8f acc;
#pragma unroll
        for (int i = 0; i < 8; ++i) acc[i] = 0.0f;
#pragma unroll
        for (int k = 0; k < 8; ++k) {
            v16bf bf = load_bfrag(Wc + (size_t)ob * CIN + k * 32, CIN, lane);
            acc = wmma_bf16(afr[k], bf, acc);
        }
        float bias = bc[ob + col];
#pragma unroll
        for (int r = 0; r < 8; ++r) {
            int pix = pbase + r + rhi;
            yb[(size_t)pix * COUT + ob + col] = (__bf16)(acc[r] + bias);
        }
    }
}

// =====================================================================
// Kernel 3: fused flash attention + residual.
// Block: 256 thr (8 waves), 128 queries/block, loop over 64-key tiles.
// LDS: Vt [256][72] bf16 (36864 B) + P scratch 8x[16][72] bf16 (18432 B)
//      reused as O-transpose [64][132] f32 (33792 B).  55296 B total.
// =====================================================================
#define VPAD 72
#define OPAD 132
__global__ void attn_kernel(const __bf16* __restrict__ y,
                            const float* __restrict__ feat,
                            const float* __restrict__ gmm,
                            float* __restrict__ out) {
    extern __shared__ char smem[];
    __bf16* vs = (__bf16*)smem;                          // [256][VPAD]
    __bf16* ps = (__bf16*)(smem + 256 * VPAD * 2);       // 8 x [16][VPAD]
    float*  os = (float*)smem;                           // [64][OPAD] (reuse)

    int blk = blockIdx.x;
    int b  = blk >> 5;
    int q0 = (blk & 31) * 128;
    int tid = threadIdx.x, wave = tid >> 5, lane = tid & 31;
    int col = lane & 15, rhi = (lane >> 4) * 8;

    const __bf16* yb = y + (size_t)b * NPIX * COUT;
    int qbase = q0 + wave * 16;

    // Q A-fragment: 16 queries x 32 channels (resident for whole kernel)
    v16bf qfr = load_afrag(yb + (size_t)qbase * COUT + QOFF, COUT, lane);

    v8f m, l, O[16];
#pragma unroll
    for (int i = 0; i < 8; ++i) { m[i] = -__builtin_inff(); l[i] = 0.0f; }
#pragma unroll
    for (int t = 0; t < 16; ++t)
#pragma unroll
        for (int i = 0; i < 8; ++i) O[t][i] = 0.0f;

    __bf16* pw = ps + wave * 16 * VPAD;

    for (int kt = 0; kt < 64; ++kt) {
        int key0 = kt * 64;
        __syncthreads();
        // ---- stage V tile transposed: vs[ch][key] ----
#pragma unroll
        for (int i = 0; i < 8; ++i) {
            int idx = i * 256 + tid;                     // 0..2047
            int key = idx >> 5, c8 = (idx & 31) * 8;
            v8bf vv = *(const v8bf*)(yb + (size_t)(key0 + key) * COUT + VOFF + c8);
#pragma unroll
            for (int j = 0; j < 8; ++j) vs[(c8 + j) * VPAD + key] = vv[j];
        }
        __syncthreads();
        if (kt + 1 < 64)   // gfx1250 global_prefetch_b8: pull next key tile toward L2
            __builtin_prefetch(yb + (size_t)(key0 + 64 + (tid & 63)) * COUT + VOFF, 0, 1);

        // ---- S = Q * K^T (4 WMMAs, K read straight from L2-resident y) ----
        v8f S[4];
#pragma unroll
        for (int s = 0; s < 4; ++s) {
            v16bf kf = load_bfrag(yb + (size_t)(key0 + s * 16) * COUT + KOFF, COUT, lane);
            v8f z;
#pragma unroll
            for (int i = 0; i < 8; ++i) z[i] = 0.0f;
            S[s] = wmma_bf16(qfr, kf, z);
        }

        // ---- online softmax (row stats live on VGPR components) ----
        v8f t;
#pragma unroll
        for (int i = 0; i < 8; ++i)
            t[i] = fmaxf(fmaxf(S[0][i], S[1][i]), fmaxf(S[2][i], S[3][i]));
#pragma unroll
        for (int d = 1; d < 16; d <<= 1)
#pragma unroll
            for (int i = 0; i < 8; ++i)
                t[i] = fmaxf(t[i], __shfl_xor(t[i], d, 32));
        v8f sc;
#pragma unroll
        for (int i = 0; i < 8; ++i) {
            float mn = fmaxf(m[i], t[i]);
            sc[i] = __expf(m[i] - mn);
            m[i] = mn;
        }
        v8f rs;
#pragma unroll
        for (int i = 0; i < 8; ++i) rs[i] = 0.0f;
#pragma unroll
        for (int s = 0; s < 4; ++s)
#pragma unroll
            for (int i = 0; i < 8; ++i) {
                float p = __expf(S[s][i] - m[i]);
                S[s][i] = p; rs[i] += p;
            }
#pragma unroll
        for (int d = 1; d < 16; d <<= 1)
#pragma unroll
            for (int i = 0; i < 8; ++i) rs[i] += __shfl_xor(rs[i], d, 32);
#pragma unroll
        for (int i = 0; i < 8; ++i) l[i] = l[i] * sc[i] + rs[i];
#pragma unroll
        for (int ct = 0; ct < 16; ++ct)
#pragma unroll
            for (int i = 0; i < 8; ++i) O[ct][i] *= sc[i];

        // ---- P: C-layout -> row-major bf16 in per-wave LDS scratch ----
#pragma unroll
        for (int s = 0; s < 4; ++s)
#pragma unroll
            for (int i = 0; i < 8; ++i)
                pw[(i + rhi) * VPAD + s * 16 + col] = (__bf16)S[s][i];

        // ---- O += P * V  (2 k-steps x 16 channel tiles = 32 WMMAs) ----
#pragma unroll
        for (int h = 0; h < 2; ++h) {
            v16bf pf = load_afrag(pw + h * 32, VPAD, lane);
#pragma unroll
            for (int ct = 0; ct < 16; ++ct) {
                v16bf vf = load_bfrag(vs + (size_t)(ct * 16) * VPAD + h * 32, VPAD, lane);
                O[ct] = wmma_bf16(pf, vf, O[ct]);
            }
        }
    }

    // ---- epilogue: O/l, transpose via LDS, out = gamma*O + feat ----
    float g = gmm[0];
    v8f rl;
#pragma unroll
    for (int i = 0; i < 8; ++i) rl[i] = 1.0f / l[i];

    const float* fb = feat + (size_t)b * CIN * NPIX;
    float* ob = out + (size_t)b * CIN * NPIX;

    for (int chunk = 0; chunk < 4; ++chunk) {            // 64 channels per chunk
        __syncthreads();
#pragma unroll
        for (int ct = 0; ct < 4; ++ct) {
            int ctg = chunk * 4 + ct;
            int chl = ct * 16 + col;                     // 0..63
#pragma unroll
            for (int r = 0; r < 8; ++r)
                os[chl * OPAD + (wave * 16 + r + rhi)] = O[ctg][r] * rl[r];
        }
        __syncthreads();
#pragma unroll 4
        for (int i = 0; i < 32; ++i) {
            int idx = i * 256 + tid;                     // 0..8191
            int ch = idx >> 7, qq = idx & 127;
            size_t a = (size_t)(chunk * 64 + ch) * NPIX + q0 + qq;
            ob[a] = g * os[ch * OPAD + qq] + fb[a];      // coalesced
        }
    }
}

// =====================================================================
// launcher
// =====================================================================
extern "C" void kernel_launch(void* const* d_in, const int* in_sizes, int n_in,
                              void* d_out, int out_size, void* d_ws, size_t ws_size,
                              hipStream_t stream) {
    const float* feat = (const float*)d_in[0];
    const float* w1 = (const float*)d_in[1];
    const float* b1 = (const float*)d_in[2];
    const float* w2 = (const float*)d_in[3];
    const float* b2 = (const float*)d_in[4];
    const float* w3 = (const float*)d_in[5];
    const float* b3 = (const float*)d_in[6];
    const float* gm = (const float*)d_in[7];
    float* out = (float*)d_out;

    char* ws = (char*)d_ws;
    __bf16* Wc = (__bf16*)ws;                        // 320*256 bf16 = 163840 B
    float*  bc = (float*)(ws + 163840);              // 320 f32
    __bf16* yq = (__bf16*)(ws + 165376);             // [8][4096][320] bf16 ~ 21 MB

    pack_kernel<<<320, 256, 0, stream>>>(w1, b1, w2, b2, w3, b3, Wc, bc);
    proj_kernel<<<512, 256, 64 * XPAD * 2, stream>>>(feat, Wc, bc, yq);
    attn_kernel<<<256, 256, 256 * VPAD * 2 + 8 * 16 * VPAD * 2, stream>>>(yq, feat, gm, out);
}
